// AttnBlock_60687887893133
// MI455X (gfx1250) — matrix operand order repaired and checked by reading, not compiled, not run on hardware
//
#include <hip/hip_runtime.h>
#include <hip/hip_bf16.h>
#include <cstdint>
#include <cstddef>

// ---- CDNA5 WMMA types -------------------------------------------------------
typedef __attribute__((ext_vector_type(16))) __bf16 v16bf;
typedef __attribute__((ext_vector_type(8)))  float  v8f;

#define DEV static __device__ __forceinline__

// fp32 -> bf16 round-to-nearest-even
DEV unsigned short f2bf(float f) {
    unsigned int u = __float_as_uint(f);
    u += 0x7fffu + ((u >> 16) & 1u);
    return (unsigned short)(u >> 16);
}

union Frag {
    v16bf v;
    uint4 q[2];
};

// A-matrix fragment (16x32 bf16, M x K). Lane L: row M = L&15 (caller bakes the
// row into rowptr), K values = k32 + base + {0..7} and k32 + base + {16..23},
// base = (L>>4)*8.  (ISA 7.12.2 "16-bit A-Matrix 16x32")
DEV Frag load_frag_A(const unsigned short* rowptr, int lane, int k32) {
    Frag f;
    const int h0 = k32 + ((lane >> 4) << 3);
    f.q[0] = *(const uint4*)(rowptr + h0);
    f.q[1] = *(const uint4*)(rowptr + h0 + 16);
    return f;
}

// B-matrix fragment (32x16 bf16, K x N). Lane L: col N = L&15 (baked into
// rowptr), K values = k32 + base + {0..7} and k32 + base + {16..23}, base = (L>>4)*8.
DEV Frag load_frag_B(const unsigned short* rowptr, int lane, int k32) {
    Frag f;
    const int h0 = k32 + ((lane >> 4) << 3);
    f.q[0] = *(const uint4*)(rowptr + h0);
    f.q[1] = *(const uint4*)(rowptr + h0 + 16);
    return f;
}

DEV v8f wmma_bf16(const Frag& a, const Frag& b, v8f c) {
    // (neg_a, A, neg_b, B, c_mod, C, reuse_a, reuse_b)
    return __builtin_amdgcn_wmma_f32_16x16x32_bf16(false, a.v, false, b.v,
                                                   (short)0, c, false, false);
}

// Async memory->LDS copy, 16 bytes per lane. Tracked by ASYNCcnt.
// LDS address = low 32 bits of the shared-memory pointer (LDS aperture rule).
DEV void async_b128(void* lds, const void* gaddr) {
    const unsigned l = (unsigned)(uintptr_t)lds;
    asm volatile("global_load_async_to_lds_b128 %0, %1, off"
                 :: "v"(l), "v"(gaddr) : "memory");
}
DEV void wait_asynccnt0() {
    asm volatile("s_wait_asynccnt 0x0" ::: "memory");
}

// Problem sizes
constexpr int CB = 8;       // batch
constexpr int CC = 128;     // channels
constexpr int CN = 4096;    // H*W

// ---- Kernel 1: weights -> bf16, transposed Wt[mat][d][c] = W[mat][c][d] -----
__global__ void wconv_kernel(const float* __restrict__ W0, const float* __restrict__ W1,
                             const float* __restrict__ W2, const float* __restrict__ W3,
                             unsigned short* __restrict__ Wt) {
    int idx = blockIdx.x * blockDim.x + threadIdx.x;   // 4*128*128 = 65536
    int mat  = idx >> 14;
    int rest = idx & 16383;
    int c = rest >> 7, d = rest & 127;
    const float* Ws = (mat == 0) ? W0 : (mat == 1) ? W1 : (mat == 2) ? W2 : W3;
    Wt[(mat << 14) + (d << 7) + c] = f2bf(Ws[(c << 7) + d]);
}

// ---- Kernel 2: x[b][c][n] -> Xt[b][n][c] bf16 -------------------------------
__global__ void xconv_kernel(const float* __restrict__ x, unsigned short* __restrict__ Xt) {
    int idx = blockIdx.x * blockDim.x + threadIdx.x;   // 8*128*4096
    int b   = idx >> 19;
    int rem = idx & ((1 << 19) - 1);
    int c   = rem >> 12;
    int n   = rem & 4095;
    Xt[((size_t)(b * CN + n)) * CC + c] = f2bf(x[idx]);
}

// ---- Kernel 3: QKV projections: D[d][n] = sum_c Wt[d][c] * Xt[n][c] ---------
// Q stored [B][N][C], K stored [B][N][C], V stored [B][C][N].
__global__ void __launch_bounds__(256)
qkv_kernel(const unsigned short* __restrict__ Xt, const unsigned short* __restrict__ Wt,
           const float* __restrict__ b0, const float* __restrict__ b1,
           const float* __restrict__ b2,
           unsigned short* __restrict__ Qbf, unsigned short* __restrict__ Kbf,
           unsigned short* __restrict__ Vbf) {
    const int tid  = threadIdx.x;
    const int lane = tid & 31, wave = tid >> 5;
    const int gtile = blockIdx.x * 8 + wave;           // 0..2047 (B * N/16)
    const int b  = gtile >> 8;
    const int n0 = (gtile & 255) << 4;

    // B-fragments of X^T: lane col n = n0 + (lane&15), contiguous c
    Frag xf[4];
    const unsigned short* xrow = Xt + ((size_t)(b * CN + n0 + (lane & 15))) * CC;
    #pragma unroll
    for (int k = 0; k < 4; ++k) xf[k] = load_frag_B(xrow, lane, 32 * k);

    const float* biases[3] = {b0, b1, b2};
    #pragma unroll
    for (int mat = 0; mat < 3; ++mat) {
        const unsigned short* Wm = Wt + (mat << 14);
        const float* bias = biases[mat];
        #pragma unroll
        for (int dt = 0; dt < 8; ++dt) {
            const unsigned short* arow = Wm + (((dt << 4) + (lane & 15)) << 7);
            // Preload all A-fragments, then issue WMMAs back-to-back.
            Frag af[4];
            #pragma unroll
            for (int k = 0; k < 4; ++k) af[k] = load_frag_A(arow, lane, 32 * k);
            v8f acc = {};
            #pragma unroll
            for (int k = 0; k < 4; ++k) acc = wmma_bf16(af[k], xf[k], acc);
            #pragma unroll
            for (int r = 0; r < 8; ++r) {
                const int d = (dt << 4) + r + ((lane >> 4) << 3);
                const int n = n0 + (lane & 15);
                const unsigned short hv = f2bf(acc[r] + bias[d]);
                if (mat == 0)      Qbf[((size_t)(b * CN + n)) * CC + d] = hv;
                else if (mat == 1) Kbf[((size_t)(b * CN + n)) * CC + d] = hv;
                else               Vbf[((size_t)(b * CC + d)) * CN + n] = hv;
            }
        }
    }
}

// ---- Kernel 4: flash attention ---------------------------------------------
// Block = 8 waves; each block = one (b, 128-row query chunk); wave w owns a
// 16-row query tile. Loop over key/value chunks of 128, staged to LDS with
// GLOBAL_LOAD_ASYNC_TO_LDS (ASYNCcnt), next chunk prefetched into L2.
__global__ void __launch_bounds__(256)
attn_kernel(const unsigned short* __restrict__ Qbf, const unsigned short* __restrict__ Kbf,
            const unsigned short* __restrict__ Vbf, unsigned short* __restrict__ Hbf) {
    __shared__ __align__(16) unsigned short Kt[128 * 128];   // [m][c] 32 KB
    __shared__ __align__(16) unsigned short Vt[128 * 128];   // [c][m] 32 KB
    __shared__ __align__(16) unsigned short Sp[8 * 16 * 128]; // per-wave P tile

    const int tid  = threadIdx.x;
    const int lane = tid & 31, wave = tid >> 5;
    const int b  = blockIdx.x >> 5;
    const int n0 = (blockIdx.x & 31) << 7;
    const float scale = 0.0883883476f;                 // 1/sqrt(128)

    // Q A-fragments: lane row n = qbase + (lane&15), contiguous c per row
    Frag qf[4];
    const unsigned short* qp = Qbf + ((size_t)(b * CN + n0 + (wave << 4) + (lane & 15))) * CC;
    #pragma unroll
    for (int k = 0; k < 4; ++k) qf[k] = load_frag_A(qp, lane, 32 * k);

    float rowm[8], rowl[8];
    v8f o[8];
    const v8f vzero = {};
    #pragma unroll
    for (int r = 0; r < 8; ++r) { rowm[r] = -1e30f; rowl[r] = 0.f; o[r] = vzero; }

    unsigned short* sp = Sp + (wave << 11);            // 16x128 bf16 scratch

    // Per-thread staging coordinates (flat 0..2047 in 8 steps of 256)
    const int srow0 = tid >> 4;                        // +16 per step
    const int sc16  = (tid & 15) << 3;                 // half-offset, 16B chunk

    for (int m0 = 0; m0 < CN; m0 += 128) {
        __syncthreads();
        // Stage K chunk -> Kt[m][c], V chunk -> Vt[c][m] via async-to-LDS.
        #pragma unroll
        for (int i = 0; i < 8; ++i) {
            const int row = srow0 + (i << 4);
            async_b128(Kt + (row << 7) + sc16,
                       Kbf + ((size_t)(b * CN + m0 + row)) * CC + sc16);
            async_b128(Vt + (row << 7) + sc16,
                       Vbf + ((size_t)(b * CC + row)) * CN + m0 + sc16);
        }
        wait_asynccnt0();
        __syncthreads();

        // Prefetch next chunk into L2 while we compute on this one.
        if (m0 + 128 < CN) {
            __builtin_prefetch(Kbf + ((size_t)(b * CN + m0 + 128 + (tid >> 1))) * CC +
                                   ((tid & 1) << 6), 0, 1);
            __builtin_prefetch(Vbf + ((size_t)(b * CC + (tid >> 1))) * CN + m0 + 128 +
                                   ((tid & 1) << 6), 0, 1);
        }

        // S = (Q K^T) * scale for 8 m-subtiles
        v8f s[8];
        #pragma unroll
        for (int ms = 0; ms < 8; ++ms) {
            const unsigned short* krow = Kt + (((ms << 4) + (lane & 15)) << 7);
            Frag kf[4];
            #pragma unroll
            for (int k = 0; k < 4; ++k) kf[k] = load_frag_B(krow, lane, 32 * k);
            v8f acc = {};
            #pragma unroll
            for (int k = 0; k < 4; ++k) acc = wmma_bf16(qf[k], kf[k], acc);
            #pragma unroll
            for (int r = 0; r < 8; ++r) s[ms][r] = acc[r] * scale;
        }

        // Online softmax: row n = r + 8*(lane>>4); its 16 m-values per subtile
        // live in one register slot across a 16-lane half -> shfl_xor(<16).
        #pragma unroll
        for (int r = 0; r < 8; ++r) {
            float cm = s[0][r];
            #pragma unroll
            for (int ms = 1; ms < 8; ++ms) cm = fmaxf(cm, s[ms][r]);
            #pragma unroll
            for (int off = 1; off < 16; off <<= 1) cm = fmaxf(cm, __shfl_xor(cm, off, 32));
            const float mnew  = fmaxf(rowm[r], cm);
            const float alpha = __expf(rowm[r] - mnew);
            rowm[r] = mnew;
            rowl[r] *= alpha;
            #pragma unroll
            for (int ct = 0; ct < 8; ++ct) o[ct][r] *= alpha;
        }

        // P = exp(S - m), bounce through LDS to re-layout D-frag -> A-frag
        float psum[8];
        #pragma unroll
        for (int r = 0; r < 8; ++r) psum[r] = 0.f;
        #pragma unroll
        for (int ms = 0; ms < 8; ++ms) {
            #pragma unroll
            for (int r = 0; r < 8; ++r) {
                const float p = __expf(s[ms][r] - rowm[r]);
                psum[r] += p;
                sp[((r + ((lane >> 4) << 3)) << 7) + (ms << 4) + (lane & 15)] = f2bf(p);
            }
        }
        #pragma unroll
        for (int r = 0; r < 8; ++r) {
            float ps = psum[r];
            #pragma unroll
            for (int off = 1; off < 16; off <<= 1) ps += __shfl_xor(ps, off, 32);
            rowl[r] += ps;
        }

        // O += P * V
        const unsigned short* prow = sp + ((lane & 15) << 7);
        #pragma unroll
        for (int k = 0; k < 4; ++k) {
            Frag pf = load_frag_A(prow, lane, 32 * k);
            #pragma unroll
            for (int ct = 0; ct < 8; ++ct) {
                const unsigned short* vrow = Vt + (((ct << 4) + (lane & 15)) << 7);
                Frag vf = load_frag_B(vrow, lane, 32 * k);
                o[ct] = wmma_bf16(pf, vf, o[ct]);
            }
        }
    }

    // Normalize and store H[b][n][c] (bf16)
    #pragma unroll
    for (int r = 0; r < 8; ++r) {
        const float rinv = 1.0f / rowl[r];
        const int n = n0 + (wave << 4) + r + ((lane >> 4) << 3);
        #pragma unroll
        for (int ct = 0; ct < 8; ++ct) {
            const int c = (ct << 4) + (lane & 15);
            Hbf[((size_t)(b * CN + n)) * CC + c] = f2bf(o[ct][r] * rinv);
        }
    }
}

// ---- Kernel 5: out = x + b3 + W3^T h ---------------------------------------
__global__ void __launch_bounds__(256)
proj_kernel(const float* __restrict__ x, const unsigned short* __restrict__ Hbf,
            const unsigned short* __restrict__ Wt3, const float* __restrict__ b3,
            float* __restrict__ out) {
    const int tid  = threadIdx.x;
    const int lane = tid & 31, wave = tid >> 5;
    const int gtile = blockIdx.x * 8 + wave;
    const int b  = gtile >> 8;
    const int n0 = (gtile & 255) << 4;

    Frag hf[4];
    const unsigned short* hrow = Hbf + ((size_t)(b * CN + n0 + (lane & 15))) * CC;
    #pragma unroll
    for (int k = 0; k < 4; ++k) hf[k] = load_frag_B(hrow, lane, 32 * k);

    #pragma unroll
    for (int dt = 0; dt < 8; ++dt) {
        const unsigned short* arow = Wt3 + (((dt << 4) + (lane & 15)) << 7);
        Frag af[4];
        #pragma unroll
        for (int k = 0; k < 4; ++k) af[k] = load_frag_A(arow, lane, 32 * k);
        v8f acc = {};
        #pragma unroll
        for (int k = 0; k < 4; ++k) acc = wmma_bf16(af[k], hf[k], acc);
        #pragma unroll
        for (int r = 0; r < 8; ++r) {
            const int d = (dt << 4) + r + ((lane >> 4) << 3);
            const int n = n0 + (lane & 15);
            const size_t idx = ((size_t)(b * CC + d)) * CN + n;
            out[idx] = x[idx] + b3[d] + acc[r];
        }
    }
}

// ---- Host launcher ----------------------------------------------------------
extern "C" void kernel_launch(void* const* d_in, const int* in_sizes, int n_in,
                              void* d_out, int out_size, void* d_ws, size_t ws_size,
                              hipStream_t stream) {
    (void)in_sizes; (void)n_in; (void)out_size; (void)ws_size;
    const float* x  = (const float*)d_in[0];
    const float* W0 = (const float*)d_in[1];
    const float* b0 = (const float*)d_in[2];
    const float* W1 = (const float*)d_in[3];
    const float* b1 = (const float*)d_in[4];
    const float* W2 = (const float*)d_in[5];
    const float* b2 = (const float*)d_in[6];
    const float* W3 = (const float*)d_in[7];
    const float* b3 = (const float*)d_in[8];
    float* out = (float*)d_out;

    char* ws = (char*)d_ws;
    const size_t SZW = (size_t)4 * 128 * 128 * 2;          // 128 KB
    const size_t SZT = (size_t)CB * CN * CC * 2;           // 8 MB each
    unsigned short* Wt  = (unsigned short*)(ws);
    unsigned short* Xt  = (unsigned short*)(ws + SZW);
    unsigned short* Qbf = (unsigned short*)(ws + SZW + 1 * SZT);
    unsigned short* Kbf = (unsigned short*)(ws + SZW + 2 * SZT);
    unsigned short* Vbf = (unsigned short*)(ws + SZW + 3 * SZT);
    unsigned short* Hbf = (unsigned short*)(ws + SZW + 4 * SZT);

    hipLaunchKernelGGL(wconv_kernel, dim3(256),   dim3(256), 0, stream, W0, W1, W2, W3, Wt);
    hipLaunchKernelGGL(xconv_kernel, dim3(16384), dim3(256), 0, stream, x, Xt);
    hipLaunchKernelGGL(qkv_kernel,   dim3(256),   dim3(256), 0, stream,
                       Xt, Wt, b0, b1, b2, Qbf, Kbf, Vbf);
    hipLaunchKernelGGL(attn_kernel,  dim3(256),   dim3(256), 0, stream, Qbf, Kbf, Vbf, Hbf);
    hipLaunchKernelGGL(proj_kernel,  dim3(256),   dim3(256), 0, stream,
                       x, Hbf, Wt + 3 * 16384, b3, out);
}
